// FeatureAlign_V2_77653008712239
// MI455X (gfx1250) — matrix operand order repaired and checked by reading, not compiled, not run on hardware
//
#include <hip/hip_runtime.h>
#include <hip/hip_bf16.h>

typedef __attribute__((ext_vector_type(16))) __bf16 bf16x16;
typedef __attribute__((ext_vector_type(8)))  float  f32x8;

struct U8 { unsigned int x[8]; };

__device__ __forceinline__ unsigned short f2bf(float f) {
    unsigned int u = __float_as_uint(f);
    unsigned int r = u + 0x7FFFu + ((u >> 16) & 1u);   // round-to-nearest-even
    return (unsigned short)(r >> 16);
}

// ---------------------------------------------------------------------------
// Support kernels
// ---------------------------------------------------------------------------

__global__ __launch_bounds__(256) void pool_k(const float* __restrict__ fl,
                                              float* __restrict__ aa) {
    __shared__ float red[256];
    int bc = blockIdx.x;                       // b*320 + c
    const float* src = fl + (long)bc * 4096;
    float s = 0.f;
    for (int i = threadIdx.x; i < 4096; i += 256) s += src[i];
    red[threadIdx.x] = s;
    __syncthreads();
    for (int off = 128; off > 0; off >>= 1) {
        if (threadIdx.x < off) red[threadIdx.x] += red[threadIdx.x + off];
        __syncthreads();
    }
    if (threadIdx.x == 0) aa[bc] = red[0] * (1.f / 4096.f);
}

__global__ __launch_bounds__(256) void atten_k(const float* __restrict__ w,
                                               const float* __restrict__ aa,
                                               float* __restrict__ scale) {
    int i = blockIdx.x * 256 + threadIdx.x;
    if (i >= 640) return;
    int b = i / 320, o = i - b * 320;
    const float* av = aa + b * 320;
    const float* wr = w + (long)o * 320;
    float s = 0.f;
    for (int c = 0; c < 320; ++c) s += wr[c] * av[c];
    scale[i] = 1.f + 1.f / (1.f + __expf(-s));
}

__global__ __launch_bounds__(256) void xbf_k(const float* __restrict__ fl,
                                             const float* __restrict__ scale,
                                             unsigned short* __restrict__ xb) {
    long i = (long)blockIdx.x * 256 + threadIdx.x;   // 2*320*4096
    long bc = i >> 12;
    xb[i] = f2bf(fl[i] * scale[bc]);
}

__global__ __launch_bounds__(256) void cvt_k(const float* __restrict__ s,
                                             unsigned short* __restrict__ d, long n) {
    long i = (long)blockIdx.x * 256 + threadIdx.x;
    if (i < n) d[i] = f2bf(s[i]);
}

// bilinear 32x32 -> 64x64 (half-pixel). Writes up (f32) and aaa rows [512,1024) bf16 = 2*up.
__global__ __launch_bounds__(256) void upsample_k(const float* __restrict__ fs,
                                                  float* __restrict__ up,
                                                  unsigned short* __restrict__ aaa) {
    long i = (long)blockIdx.x * 256 + threadIdx.x;   // 2*512*4096
    int n = (int)(i & 4095);
    long bc = i >> 12;                               // b*512 + c
    int y = n >> 6, x = n & 63;
    float sy = 0.5f * y - 0.25f;
    float sx = 0.5f * x - 0.25f;
    float y0f = floorf(sy), x0f = floorf(sx);
    float fy = sy - y0f, fx = sx - x0f;
    int y0 = (int)y0f, x0 = (int)x0f;
    int y0c = min(max(y0, 0), 31), y1c = min(max(y0 + 1, 0), 31);
    int x0c = min(max(x0, 0), 31), x1c = min(max(x0 + 1, 0), 31);
    const float* src = fs + bc * 1024;
    float v00 = src[y0c * 32 + x0c], v01 = src[y0c * 32 + x1c];
    float v10 = src[y1c * 32 + x0c], v11 = src[y1c * 32 + x1c];
    float v0 = v00 + fx * (v01 - v00);
    float v1 = v10 + fx * (v11 - v10);
    float v  = v0  + fy * (v1  - v0);
    up[i] = v;
    long b = bc >> 9, c = bc & 511;
    aaa[b * (1024L * 4096) + (512 + c) * 4096L + n] = f2bf(2.f * v);
}

// deformable bilinear gather -> cols bf16 [B, 4608, 4096], row index = c_glob*9 + k
__global__ __launch_bounds__(256) void cols_k(const float* __restrict__ om,   // [B,216,4096]
                                              const float* __restrict__ up,   // [B,512,4096]
                                              unsigned short* __restrict__ cols) {
    int n  = blockIdx.x * 256 + threadIdx.x;   // 0..4095
    int dk = blockIdx.y;                       // dg*9 + k, 0..71
    int b  = blockIdx.z;
    int dg = dk / 9, k = dk - dg * 9;
    int ky = k / 3 - 1, kx = k - (k / 3) * 3 - 1;
    long omB = (long)b * 216 * 4096;
    float dy = om[omB + (long)dk * 4096 + n];
    float dx = om[omB + (long)(72 + dk) * 4096 + n];
    float ms = om[omB + (long)(144 + dk) * 4096 + n];
    ms = 1.f / (1.f + __expf(-ms));
    int y = n >> 6, x = n & 63;
    float py = (float)(y + ky) + dy;
    float px = (float)(x + kx) + dx;
    float y0f = floorf(py), x0f = floorf(px);
    float wy = py - y0f, wx = px - x0f;
    int y0 = (int)y0f, x0 = (int)x0f;
    int y1 = y0 + 1, x1 = x0 + 1;
    bool vy0 = (y0 >= 0 && y0 < 64), vy1 = (y1 >= 0 && y1 < 64);
    bool vx0 = (x0 >= 0 && x0 < 64), vx1 = (x1 >= 0 && x1 < 64);
    int y0c = min(max(y0, 0), 63), y1c = min(max(y1, 0), 63);
    int x0c = min(max(x0, 0), 63), x1c = min(max(x1, 0), 63);
    float w00 = (vy0 && vx0) ? (1.f - wy) * (1.f - wx) : 0.f;
    float w01 = (vy0 && vx1) ? (1.f - wy) * wx : 0.f;
    float w10 = (vy1 && vx0) ? wy * (1.f - wx) : 0.f;
    float w11 = (vy1 && vx1) ? wy * wx : 0.f;
    int i00 = y0c * 64 + x0c, i01 = y0c * 64 + x1c;
    int i10 = y1c * 64 + x0c, i11 = y1c * 64 + x1c;
    long upB  = (long)b * 512 * 4096;
    long colB = (long)b * 4608L * 4096;
    for (int c = 0; c < 64; ++c) {
        int ch = dg * 64 + c;
        const float* f = up + upB + (long)ch * 4096;
        float v = w00 * f[i00] + w01 * f[i01] + w10 * f[i10] + w11 * f[i11];
        v *= ms;
        cols[colB + (long)(ch * 9 + k) * 4096 + n] = f2bf(v);
    }
}

// ---------------------------------------------------------------------------
// bf16 WMMA GEMM: C[M, 4096] = A[M,K] x B[K,4096]  (per-batch B/C strides)
// Block tile 128x128, BK=32, 8 waves; wave w owns rows [16w,16w+16) x 128 cols.
// Register + LDS double buffering: next tile's global loads issue before the
// WMMA phase and are committed to the alternate LDS buffer after it; one
// barrier per k-step. B fragments pipelined through 2 register buffers.
// convMode: branchless 3x3 im2col gather (zero pad) from [512,64,64] bf16.
// ---------------------------------------------------------------------------
#define BM 128
#define BN 128
#define BKK 32
#define LDSS 40   // LDS row stride in elements (80 B, multiple of 16 B)

__global__ __launch_bounds__(256)
void gemm_bf16_wmma(const unsigned short* __restrict__ A,    // [M,K]
                    const unsigned short* __restrict__ Bsrc, // [K,4096] or conv src [512,4096]
                    long bBatch,
                    float* __restrict__ outF, long outFBatch,
                    unsigned short* __restrict__ outBf, long outBfBatch,
                    const float* __restrict__ bias,
                    const float* __restrict__ resid, long residBatch,
                    int M, int Kdim, int relu, int convMode) {
    __shared__ unsigned short As[2][BM * LDSS];   // 2 x 10240 B
    __shared__ unsigned short Bs[2][BN * LDSS];   // 2 x 10240 B, transposed: Bs[n][k]
    const int N = 4096;
    int tid  = threadIdx.x;
    int wave = tid >> 5, lane = tid & 31;
    int half = lane >> 4, l15 = lane & 15;
    int m0 = blockIdx.y * BM;
    int n0 = blockIdx.x * BN;
    int b  = blockIdx.z;
    const unsigned short* Bp = Bsrc + (long)b * bBatch;

    // A staging map: thread covers row rA, 16 cols starting at cA (2 x uint4)
    int rA = tid >> 1;
    int cA = (tid & 1) * 16;
    bool aValid = (m0 + rA) < M;
    const unsigned short* aRow = A + (long)(m0 + rA) * Kdim + cA;
    // B staging map: thread covers k-row klB, 16 cols starting at nlB
    int klB = tid >> 3;
    int nlB = (tid & 7) * 16;
    int nbase = n0 + nlB;

    // ---- staged tile registers ----
    uint4 a0 = make_uint4(0, 0, 0, 0), a1 = make_uint4(0, 0, 0, 0);
    uint4 b0 = make_uint4(0, 0, 0, 0), b1 = make_uint4(0, 0, 0, 0);
    unsigned short raw[16];
    int cvXb = 0;
    bool cvYv = false;

    // load tile at k-offset kt into the staged registers
    auto loadTile = [&](int kt) {
        if (aValid) {
            const uint4* p = (const uint4*)(aRow + kt);
            a0 = p[0];
            a1 = p[1];
            if (kt + 2 * BKK < Kdim)
                __builtin_prefetch(aRow + kt + 2 * BKK, 0, 1);
        }
        if (convMode) {
            int kk = kt + klB;
            int c  = kk / 9;
            int r9 = kk - c * 9;
            int t3 = r9 / 3;
            int ky = t3 - 1, kx = r9 - t3 * 3 - 1;
            int y  = (nbase >> 6) + ky;               // constant over the 16-chunk
            int yc = min(max(y, 0), 63);
            cvYv = (unsigned)y < 64u;
            cvXb = (nbase & 63) + kx;
            const unsigned short* rowp = Bp + (long)c * 4096 + yc * 64;
            #pragma unroll
            for (int j = 0; j < 16; ++j) {
                int xc = min(max(cvXb + j, 0), 63);
                raw[j] = rowp[xc];                    // always in-bounds
            }
        } else {
            const unsigned short* bp = Bp + (long)(kt + klB) * N + nbase;
            b0 = ((const uint4*)bp)[0];
            b1 = ((const uint4*)bp)[1];
            if (kt + 2 * BKK < Kdim)
                __builtin_prefetch(bp + 2L * BKK * N, 0, 1);
        }
    };

    // commit the staged registers into LDS buffer `bi`
    auto storeTile = [&](int bi) {
        *(uint4*)&As[bi][rA * LDSS + cA]     = a0;
        *(uint4*)&As[bi][rA * LDSS + cA + 8] = a1;
        if (convMode) {
            #pragma unroll
            for (int j = 0; j < 16; ++j) {
                bool v = cvYv && (unsigned)(cvXb + j) < 64u;
                Bs[bi][(nlB + j) * LDSS + klB] = v ? raw[j] : (unsigned short)0;
            }
        } else {
            unsigned int w[8] = { b0.x, b0.y, b0.z, b0.w, b1.x, b1.y, b1.z, b1.w };
            #pragma unroll
            for (int j = 0; j < 8; ++j) {
                Bs[bi][(nlB + 2 * j)     * LDSS + klB] = (unsigned short)(w[j] & 0xFFFF);
                Bs[bi][(nlB + 2 * j + 1) * LDSS + klB] = (unsigned short)(w[j] >> 16);
            }
        }
    };

    f32x8 acc[8] = { {0,0,0,0,0,0,0,0}, {0,0,0,0,0,0,0,0},
                     {0,0,0,0,0,0,0,0}, {0,0,0,0,0,0,0,0},
                     {0,0,0,0,0,0,0,0}, {0,0,0,0,0,0,0,0},
                     {0,0,0,0,0,0,0,0}, {0,0,0,0,0,0,0,0} };

    // ---- prologue: stage tile 0 ----
    loadTile(0);
    storeTile(0);
    __syncthreads();

    int buf = 0;
    int rowA = wave * 16 + l15;
    for (int k0 = 0; k0 < Kdim; k0 += BKK) {
        bool hasNext = (k0 + BKK) < Kdim;
        // issue next tile's global loads before the math (latency hidden by WMMAs)
        if (hasNext) loadTile(k0 + BKK);

        // A fragment (16x32): lane row = l15; K = g*16 + half*8 + 2q (+bit)
        U8 ua;
        #pragma unroll
        for (int p = 0; p < 8; ++p) {
            int q = p & 3, g = p >> 2;
            int kk = g * 16 + half * 8 + 2 * q;
            ua.x[p] = *(const unsigned int*)&As[buf][rowA * LDSS + kk];
        }
        bf16x16 av = __builtin_bit_cast(bf16x16, ua);

        // B fragments pipelined through two register buffers
        U8 ub[2];
        #pragma unroll
        for (int p = 0; p < 8; ++p)
            ub[0].x[p] = *(const unsigned int*)&Bs[buf][l15 * LDSS + (p >> 2) * 16 + half * 8 + 2 * (p & 3)];
        #pragma unroll
        for (int nt = 0; nt < 8; ++nt) {
            if (nt < 7) {
                int col = (nt + 1) * 16 + l15;
                #pragma unroll
                for (int p = 0; p < 8; ++p)
                    ub[(nt + 1) & 1].x[p] =
                        *(const unsigned int*)&Bs[buf][col * LDSS + (p >> 2) * 16 + half * 8 + 2 * (p & 3)];
            }
            bf16x16 bv = __builtin_bit_cast(bf16x16, ub[nt & 1]);
            acc[nt] = __builtin_amdgcn_wmma_f32_16x16x32_bf16(
                          false, av, false, bv, (short)0, acc[nt], false, false);
        }

        // commit next tile into the alternate LDS buffer
        if (hasNext) storeTile(buf ^ 1);
        __syncthreads();
        buf ^= 1;
    }

    // epilogue: C layout VGPR r -> M = r + half*8 ; N = l15
    #pragma unroll
    for (int nt = 0; nt < 8; ++nt) {
        int n = n0 + nt * 16 + l15;
        #pragma unroll
        for (int r = 0; r < 8; ++r) {
            int m = m0 + wave * 16 + half * 8 + r;
            if (m < M) {
                float v = acc[nt][r];
                if (bias)  v += bias[m];
                if (relu)  v = v > 0.f ? v : 0.f;
                if (resid) v += resid[(long)b * residBatch + (long)m * N + n];
                if (outF)  outF[(long)b * outFBatch + (long)m * N + n] = v;
                if (outBf) outBf[(long)b * outBfBatch + (long)m * N + n] = f2bf(v);
            }
        }
    }
}

// ---------------------------------------------------------------------------
// Launch
// ---------------------------------------------------------------------------
extern "C" void kernel_launch(void* const* d_in, const int* in_sizes, int n_in,
                              void* d_out, int out_size, void* d_ws, size_t ws_size,
                              hipStream_t stream) {
    (void)in_sizes; (void)n_in; (void)out_size; (void)ws_size;
    const float* feat_l      = (const float*)d_in[0];  // [2,320,64,64]
    const float* feat_s      = (const float*)d_in[1];  // [2,512,32,32]
    const float* fsm_atten_w = (const float*)d_in[2];  // [320,320]
    const float* fsm_conv_w  = (const float*)d_in[3];  // [512,320]
    const float* offset_w    = (const float*)d_in[4];  // [512,1024]
    const float* dcn_om_w    = (const float*)d_in[5];  // [216,512,3,3]
    const float* dcn_om_b    = (const float*)d_in[6];  // [216]
    const float* dcn_w       = (const float*)d_in[7];  // [512,512,3,3]
    const float* dcn_b       = (const float*)d_in[8];  // [512]
    float* outp = (float*)d_out;                       // [2,512,64,64]

    char* ws = (char*)d_ws;
    size_t off = 0;
    auto alloc = [&](size_t bytes) { char* p = ws + off; off += (bytes + 255) & ~(size_t)255; return p; };

    float*          aa        = (float*)alloc(2 * 320 * 4);
    float*          scale     = (float*)alloc(2 * 320 * 4);
    unsigned short* x_bf      = (unsigned short*)alloc(2L * 320 * 4096 * 2);
    unsigned short* w_fsm_bf  = (unsigned short*)alloc(512L * 320 * 2);
    unsigned short* w_off_bf  = (unsigned short*)alloc(512L * 1024 * 2);
    unsigned short* w_om_bf   = (unsigned short*)alloc(216L * 4608 * 2);
    unsigned short* w_dcn_bf  = (unsigned short*)alloc(512L * 4608 * 2);
    float*          feat_arm  = (float*)alloc(2L * 512 * 4096 * 4);
    unsigned short* aaa       = (unsigned short*)alloc(2L * 1024 * 4096 * 2);
    float*          feat_up   = (float*)alloc(2L * 512 * 4096 * 4);
    unsigned short* off_bf    = (unsigned short*)alloc(2L * 512 * 4096 * 2);
    float*          om        = (float*)alloc(2L * 216 * 4096 * 4);
    unsigned short* cols      = (unsigned short*)alloc(2L * 4608 * 4096 * 2);

    // weight conversions f32 -> bf16
    cvt_k<<<(512L * 320  + 255) / 256, 256, 0, stream>>>(fsm_conv_w, w_fsm_bf, 512L * 320);
    cvt_k<<<(512L * 1024 + 255) / 256, 256, 0, stream>>>(offset_w,   w_off_bf, 512L * 1024);
    cvt_k<<<(216L * 4608 + 255) / 256, 256, 0, stream>>>(dcn_om_w,   w_om_bf,  216L * 4608);
    cvt_k<<<(512L * 4608 + 255) / 256, 256, 0, stream>>>(dcn_w,      w_dcn_bf, 512L * 4608);

    // FSM gate
    pool_k<<<2 * 320, 256, 0, stream>>>(feat_l, aa);
    atten_k<<<3, 256, 0, stream>>>(fsm_atten_w, aa, scale);
    xbf_k<<<(2L * 320 * 4096) / 256, 256, 0, stream>>>(feat_l, scale, x_bf);

    // upsample feat_s -> feat_up (f32) + aaa rows [512,1024) (bf16, *2)
    upsample_k<<<(2L * 512 * 4096) / 256, 256, 0, stream>>>(feat_s, feat_up, aaa);

    // GEMM1: feat_arm = fsm_conv_w @ x ; also bf16 into aaa rows [0,512)
    gemm_bf16_wmma<<<dim3(32, 4, 2), 256, 0, stream>>>(
        w_fsm_bf, x_bf, 320L * 4096,
        feat_arm, 512L * 4096, aaa, 1024L * 4096,
        nullptr, nullptr, 0, 512, 320, 0, 0);

    // GEMM2: off_feat = offset_w @ aaa  (bf16 only)
    gemm_bf16_wmma<<<dim3(32, 4, 2), 256, 0, stream>>>(
        w_off_bf, aaa, 1024L * 4096,
        nullptr, 0, off_bf, 512L * 4096,
        nullptr, nullptr, 0, 512, 1024, 0, 0);

    // GEMM3 (conv3x3 im2col-on-the-fly): om = om_w * off_feat + om_b
    gemm_bf16_wmma<<<dim3(32, 2, 2), 256, 0, stream>>>(
        w_om_bf, off_bf, 512L * 4096,
        om, 216L * 4096, nullptr, 0,
        dcn_om_b, nullptr, 0, 216, 4608, 0, 1);

    // deformable gather -> cols
    cols_k<<<dim3(16, 72, 2), 256, 0, stream>>>(om, feat_up, cols);

    // GEMM4: out = relu(dcn_w * cols + dcn_b) + feat_arm
    gemm_bf16_wmma<<<dim3(32, 4, 2), 256, 0, stream>>>(
        w_dcn_bf, cols, 4608L * 4096,
        outp, 512L * 4096, nullptr, 0,
        dcn_b, feat_arm, 512L * 4096, 512, 4608, 1, 0);
}
